// BiLSTM_CRF_15753940042183
// MI455X (gfx1250) — matrix operand order repaired and checked by reading, not compiled, not run on hardware
//
#include <hip/hip_runtime.h>
#include <hip/hip_bf16.h>

// ---------------------------------------------------------------------------
// BiLSTM-CRF for MI455X (gfx1250, wave32).
//  - Batched input projections + emission head: bf16 WMMA GEMM. Each block
//    owns one 16-row A panel staged into LDS by the Tensor Data Mover
//    (tensor_load_to_lds, double-buffered, s_wait_tensorcnt), 8 waves x
//    16x64 output tiles, v_wmma_f32_16x16x32_bf16.
//  - Recurrence: persistent 1024-thread workgroup, float4 matvec from L2.
//  - CRF: single small workgroup.
// ---------------------------------------------------------------------------

typedef __attribute__((ext_vector_type(16))) __bf16 v16bf;
typedef __attribute__((ext_vector_type(8)))  float  v8f;
typedef __attribute__((ext_vector_type(4)))  unsigned int u32x4;
typedef __attribute__((ext_vector_type(8)))  int i32x8;
typedef __attribute__((ext_vector_type(4)))  int i32x4;

#define T_LEN   2048
#define ELMO    1024
#define POSD    64
#define DIN0    (ELMO + POSD)   // 1088
#define HID     1024
#define G4H     (4 * HID)       // 4096
#define KTAG    48
#define NEGV    (-10000.0f)
#define CHUNK   64              // K-chunk staged per TDM transfer (bf16 elems)

#if defined(__has_builtin)
#if __has_builtin(__builtin_amdgcn_tensor_load_to_lds)
#define HAVE_TDM 1
#endif
#if __has_builtin(__builtin_amdgcn_s_wait_tensorcnt)
#define HAVE_TENSORCNT 1
#endif
#endif

static __device__ __forceinline__ unsigned short f32_to_bf16(float f) {
    union { float f; unsigned int u; } v; v.f = f;
    unsigned int r = v.u + 0x7FFFu + ((v.u >> 16) & 1u);   // round-to-nearest-even
    return (unsigned short)(r >> 16);
}

// ---------------------------------------------------------------------------
// Kernel 0: elementwise f32 -> bf16 (weight conversion)
// ---------------------------------------------------------------------------
__global__ void cvt_bf16_kernel(const float* __restrict__ in,
                                unsigned short* __restrict__ out, int n) {
    int idx = blockIdx.x * blockDim.x + threadIdx.x;
    if (idx < n) out[idx] = f32_to_bf16(in[idx]);
}

// ---------------------------------------------------------------------------
// Kernel 1: x0[t] = concat(sentence[t], emb_table[speech_tags[t]]) as bf16
// ---------------------------------------------------------------------------
__global__ void embed_concat_kernel(const float* __restrict__ sent,
                                    const int*   __restrict__ sp,
                                    const float* __restrict__ emb,
                                    unsigned short* __restrict__ out) {
    int idx = blockIdx.x * blockDim.x + threadIdx.x;
    if (idx >= T_LEN * DIN0) return;
    int t = idx / DIN0;
    int d = idx - t * DIN0;
    float v = (d < ELMO) ? sent[(size_t)t * ELMO + d]
                         : emb[(size_t)sp[t] * POSD + (d - ELMO)];
    out[idx] = f32_to_bf16(v);
}

// ---------------------------------------------------------------------------
// TDM staging of one A chunk: 16 rows x CHUNK bf16, row stride K elems.
// Issued by wave 0 only; completion via s_wait_tensorcnt + block barrier.
// ---------------------------------------------------------------------------
static __device__ __forceinline__
void tdm_issue_A(const unsigned short* gsrc, unsigned short* lds_dst, int K) {
#if defined(HAVE_TDM)
    if (threadIdx.x < 32) {
        unsigned long long ga = (unsigned long long)(const void*)gsrc;
        unsigned lds = (unsigned)(unsigned long long)(void*)lds_dst;
        u32x4 g0;
        g0.x = 1u;                                   // count=1, user mode
        g0.y = lds;                                  // D#.lds_addr (bytes)
        g0.z = (unsigned)(ga & 0xFFFFFFFFu);         // global_addr[31:0]
        g0.w = (unsigned)((ga >> 32) & 0x1FFFFFFu) | (2u << 30);  // addr[56:32], type=2
        unsigned td0 = CHUNK, td1 = 16;              // tensor dims (elem units)
        unsigned tl0 = CHUNK, tl1 = 16;              // tile dims
        unsigned long long s0 = (unsigned long long)K;  // dim0 stride (elems)
        i32x8 g1;
        g1[0] = (int)(1u << 16);                     // data_size=1 -> 2 bytes
        g1[1] = (int)((td0 & 0xFFFFu) << 16);        // tensor_dim0[15:0]
        g1[2] = (int)(((td0 >> 16) & 0xFFFFu) | ((td1 & 0xFFFFu) << 16));
        g1[3] = (int)(((td1 >> 16) & 0xFFFFu) | (tl0 << 16));   // tile_dim0
        g1[4] = (int)(tl1);                          // tile_dim1, tile_dim2=0
        g1[5] = (int)(s0 & 0xFFFFFFFFu);             // stride0 lo
        g1[6] = (int)((s0 >> 32) & 0xFFFFu);         // stride0 hi, stride1=0
        g1[7] = 0;
        i32x4 gz = {0, 0, 0, 0};
#if defined(__clang_major__) && (__clang_major__ >= 23)
        i32x8 gz8 = {0, 0, 0, 0, 0, 0, 0, 0};
        __builtin_amdgcn_tensor_load_to_lds(g0, g1, gz, gz, gz8, 0);
#else
        __builtin_amdgcn_tensor_load_to_lds(g0, g1, gz, gz, 0);
#endif
    }
#else
    // Fallback: cooperative copy (16 rows x 4 uint4 at 16-elem stride = 64 uint4)
    int nth = blockDim.x;
    for (int i = threadIdx.x; i < 16 * (CHUNK / 16); i += nth) {
        int row = i >> 2, seg = i & 3;
        *(uint4*)(lds_dst + row * CHUNK + seg * 16) =
            *(const uint4*)(gsrc + (size_t)row * K + seg * 16);
    }
#endif
}

static __device__ __forceinline__ void tdm_wait() {
#if defined(HAVE_TDM)
    if (threadIdx.x < 32) {
#if defined(HAVE_TENSORCNT)
        __builtin_amdgcn_s_wait_tensorcnt(0);
#else
        asm volatile("s_wait_tensorcnt 0" ::: "memory");
#endif
    }
#endif
}

// ---------------------------------------------------------------------------
// Kernel 2: WMMA GEMM  C[M,N] = A_bf16[M,K] @ Wbf16[N,K]^T + bias1 + bias2
// Block = NW waves sharing one 16-row A panel (TDM-staged in LDS, double
// buffered). Each wave owns a 16 x (16*NT) output tile: one LDS A fragment
// per K-step feeds NT back-to-back v_wmma_f32_16x16x32_bf16 ops.
// ---------------------------------------------------------------------------
union FragBF {
    v16bf v;
    uint4 q[2];
};

template <int NT>
__global__ __launch_bounds__(256)
void gemm_wmma_bf16(const unsigned short* __restrict__ A,
                    const unsigned short* __restrict__ Wb,
                    const float* __restrict__ bias1,
                    const float* __restrict__ bias2,
                    float* __restrict__ C,
                    int M, int N, int K, int ldc) {
    __shared__ __align__(16) unsigned short sA[2][16 * CHUNK];

    int nw     = blockDim.x >> 5;            // waves per block
    int wav    = threadIdx.x >> 5;
    int lane   = threadIdx.x & 31;
    int groupsN = N / (16 * NT);
    int bpr    = groupsN / nw;               // blocks per M-tile row
    int tm     = blockIdx.x / bpr;
    int tg     = (blockIdx.x - tm * bpr) * nw + wav;
    int half   = lane >> 4;                  // 0: lanes 0-15, 1: lanes 16-31
    int l15    = lane & 15;

    const unsigned short* apanel = A + (size_t)(tm * 16) * K;
    const unsigned short* wrow[NT];
    #pragma unroll
    for (int j = 0; j < NT; ++j)
        wrow[j] = Wb + (size_t)((tg * NT + j) * 16 + l15) * K;

    v8f acc[NT];
    #pragma unroll
    for (int j = 0; j < NT; ++j) acc[j] = (v8f){};

    int nc = K / CHUNK;

    // prologue: stage chunk 0
    tdm_issue_A(apanel, &sA[0][0], K);
    tdm_wait();
    __syncthreads();

    for (int c = 0; c < nc; ++c) {
        if (c + 1 < nc)
            tdm_issue_A(apanel + (c + 1) * CHUNK, &sA[(c + 1) & 1][0], K);

        const unsigned short* sbuf = &sA[c & 1][0];
        #pragma unroll
        for (int kk = 0; kk < CHUNK; kk += 32) {
            // A fragment from LDS (ISA 16-bit A 16x32 layout): row l15,
            // K = kk + half*8 + {0..7} and + 16 + {0..7}
            FragBF fa;
            int ka = kk + half * 8;
            fa.q[0] = *(const uint4*)(sbuf + l15 * CHUNK + ka);
            fa.q[1] = *(const uint4*)(sbuf + l15 * CHUNK + ka + 16);

            int kb = c * CHUNK + kk + half * 8;   // B: col l15, K = kb+{0..7} and kb+16+{0..7}
            #pragma unroll
            for (int j = 0; j < NT; ++j) {
                FragBF fb;
                fb.q[0] = *(const uint4*)(wrow[j] + kb);
                fb.q[1] = *(const uint4*)(wrow[j] + kb + 16);
                acc[j] = __builtin_amdgcn_wmma_f32_16x16x32_bf16(
                             false, fa.v, false, fb.v, (short)0, acc[j],
                             false, false);
            }
        }

        if (c + 1 < nc) tdm_wait();
        __syncthreads();
    }

    int mbase = tm * 16 + half * 8;          // C/D layout: VGPR r -> M = mbase + r
    #pragma unroll
    for (int j = 0; j < NT; ++j) {
        int n = (tg * NT + j) * 16 + l15;
        float b = (bias1 ? bias1[n] : 0.0f) + (bias2 ? bias2[n] : 0.0f);
        #pragma unroll
        for (int r = 0; r < 8; ++r)
            C[(size_t)(mbase + r) * ldc + n] = acc[j][r] + b;
    }
}

// ---------------------------------------------------------------------------
// Kernel 3: sequential LSTM recurrence, one direction.
// gates[t] = G[t] (precomputed Wih@x + biases) + Whh @ h_{t-1}
// One workgroup, 1024 threads (32 waves). h,c in LDS. Matvec: each wave owns
// 128 gate rows; per row, lanes do float4 coalesced dot + shuffle reduce.
// ---------------------------------------------------------------------------
__global__ __launch_bounds__(1024)
void lstm_recurrent_kernel(const float* __restrict__ G,
                           const float* __restrict__ Whh,
                           unsigned short* __restrict__ out_bf16,
                           int reverse, int col_off) {
    __shared__ float sh[HID];
    __shared__ float sc[HID];
    __shared__ float sy[G4H];

    int tid  = threadIdx.x;
    int wave = tid >> 5;
    int lane = tid & 31;

    sh[tid] = 0.0f;
    sc[tid] = 0.0f;
    __syncthreads();

    for (int t = 0; t < T_LEN; ++t) {
        int tt = reverse ? (T_LEN - 1 - t) : t;

        // ---- phase 1: y = Whh @ h  (rows wave*128 .. wave*128+127) ----
        int rbase = wave * 128;
        for (int rr = 0; rr < 128; ++rr) {
            int r = rbase + rr;
            const float* wr = Whh + (size_t)r * HID;
            if (rr + 1 < 128)   // warm caches for next row (global_prefetch_b8)
                __builtin_prefetch(Whh + (size_t)(r + 1) * HID + lane * 4, 0, 0);
            float p = 0.0f;
            #pragma unroll
            for (int j = 0; j < 8; ++j) {
                int k = j * 128 + lane * 4;
                float4 w = *(const float4*)(wr + k);
                float4 h = *(const float4*)(&sh[k]);
                p = fmaf(w.x, h.x, p);
                p = fmaf(w.y, h.y, p);
                p = fmaf(w.z, h.z, p);
                p = fmaf(w.w, h.w, p);
            }
            #pragma unroll
            for (int off = 16; off > 0; off >>= 1)
                p += __shfl_down(p, off, 32);
            if (lane == 0) sy[r] = p;
        }
        __syncthreads();

        // ---- phase 2: gate math, unit u = tid (PyTorch order i,f,g,o) ----
        const float* g = G + (size_t)tt * G4H;
        float ig = g[tid]            + sy[tid];
        float fg = g[HID + tid]      + sy[HID + tid];
        float gg = g[2 * HID + tid]  + sy[2 * HID + tid];
        float og = g[3 * HID + tid]  + sy[3 * HID + tid];

        float i = 1.0f / (1.0f + expf(-ig));
        float f = 1.0f / (1.0f + expf(-fg));
        float gt = tanhf(gg);
        float o = 1.0f / (1.0f + expf(-og));

        float cn = f * sc[tid] + i * gt;
        float hn = o * tanhf(cn);
        sc[tid] = cn;
        sh[tid] = hn;   // safe: all phase-1 reads completed before barrier above
        out_bf16[(size_t)tt * (2 * HID) + col_off + tid] = f32_to_bf16(hn);
        __syncthreads();
    }
}

// ---------------------------------------------------------------------------
// Kernel 4: CRF forward log-partition + gold score. One block, 64 threads.
// ---------------------------------------------------------------------------
__global__ __launch_bounds__(64)
void crf_kernel(const float* __restrict__ emis,     // [T, 48]
                const float* __restrict__ trans,    // [48, 48], trans[j*48+i]
                const int*   __restrict__ tags,
                float* __restrict__ out) {
    __shared__ float fv[KTAG];
    __shared__ float nfv[KTAG];
    __shared__ float red[64];
    __shared__ float gred[64];

    int tid = threadIdx.x;
    if (tid < KTAG) fv[tid] = (tid == 0) ? 0.0f : NEGV;   // START_IDX = 0
    __syncthreads();

    for (int t = 0; t < T_LEN; ++t) {
        if (tid < KTAG) {
            const float* tr = trans + tid * KTAG;
            float m = -3.4e38f;
            #pragma unroll 4
            for (int i = 0; i < KTAG; ++i) m = fmaxf(m, fv[i] + tr[i]);
            float s = 0.0f;
            #pragma unroll 4
            for (int i = 0; i < KTAG; ++i) s += expf(fv[i] + tr[i] - m);
            nfv[tid] = m + logf(s) + emis[(size_t)t * KTAG + tid];
        }
        __syncthreads();
        if (tid < KTAG) fv[tid] = nfv[tid];
        __syncthreads();
    }

    // terminal: logsumexp(fv + trans[END_IDX=1, :])
    red[tid] = (tid < KTAG) ? (fv[tid] + trans[1 * KTAG + tid]) : -3.4e38f;

    // gold score partials
    float gp = 0.0f;
    for (int t = tid; t < T_LEN; t += 64) {
        int prev = (t == 0) ? 0 : tags[t - 1];
        int cur  = tags[t];
        gp += trans[cur * KTAG + prev] + emis[(size_t)t * KTAG + cur];
    }
    gred[tid] = gp;
    __syncthreads();

    if (tid == 0) {
        float m = -3.4e38f;
        for (int i = 0; i < KTAG; ++i) m = fmaxf(m, red[i]);
        float s = 0.0f;
        for (int i = 0; i < KTAG; ++i) s += expf(red[i] - m);
        float fwd = m + logf(s);
        float gold = trans[1 * KTAG + tags[T_LEN - 1]];
        for (int i = 0; i < 64; ++i) gold += gred[i];
        out[0] = fwd - gold;
    }
}

// ---------------------------------------------------------------------------
// Host launcher
// ---------------------------------------------------------------------------
static inline size_t align256(size_t x) { return (x + 255) & ~(size_t)255; }

extern "C" void kernel_launch(void* const* d_in, const int* in_sizes, int n_in,
                              void* d_out, int out_size, void* d_ws, size_t ws_size,
                              hipStream_t stream) {
    (void)in_sizes; (void)n_in; (void)out_size; (void)ws_size;

    const float* sent  = (const float*)d_in[0];
    const int*   sp    = (const int*)  d_in[1];
    const int*   tags  = (const int*)  d_in[2];
    const float* emb   = (const float*)d_in[3];
    // lstm_params flattened: layer l, dir d -> base 4 + l*8 + d*4:
    //   [Wih, Whh, bih, bhh]
    const float* W_out = (const float*)d_in[20];
    const float* b_out = (const float*)d_in[21];
    const float* trans = (const float*)d_in[22];

    char* ws = (char*)d_ws;
    size_t off = 0;
    unsigned short* X0  = (unsigned short*)(ws + off); off = align256(off + (size_t)T_LEN * DIN0 * 2);
    float*          Gb  = (float*)         (ws + off); off = align256(off + (size_t)T_LEN * G4H * 4);
    unsigned short* L1  = (unsigned short*)(ws + off); off = align256(off + (size_t)T_LEN * 2 * HID * 2);
    unsigned short* L2  = (unsigned short*)(ws + off); off = align256(off + (size_t)T_LEN * 2 * HID * 2);
    float*          EMI = (float*)         (ws + off); off = align256(off + (size_t)T_LEN * KTAG * 4);
    unsigned short* WBF = (unsigned short*)(ws + off); off = align256(off + (size_t)G4H * (2 * HID) * 2);

    // 1) embedding concat -> bf16 X0
    {
        int total = T_LEN * DIN0;
        embed_concat_kernel<<<(total + 255) / 256, 256, 0, stream>>>(sent, sp, emb, X0);
    }

    // 2) per (layer, dir): weight cvt, gate GEMM, recurrence
    const int din[2] = { DIN0, 2 * HID };
    const unsigned short* xin[2] = { X0, L1 };
    unsigned short* xout[2] = { L1, L2 };

    for (int l = 0; l < 2; ++l) {
        for (int d = 0; d < 2; ++d) {
            int base = 4 + l * 8 + d * 4;
            const float* Wih = (const float*)d_in[base + 0];
            const float* Whh = (const float*)d_in[base + 1];
            const float* bih = (const float*)d_in[base + 2];
            const float* bhh = (const float*)d_in[base + 3];

            int M = T_LEN, N = G4H, K = din[l];

            int wn = N * K;
            cvt_bf16_kernel<<<(wn + 255) / 256, 256, 0, stream>>>(Wih, WBF, wn);

            // 8 waves/block, NT=4: block covers 16 x 512; grid = 128 * 8
            int groupsN = N / (16 * 4);
            int nblk = (M >> 4) * (groupsN / 8);
            gemm_wmma_bf16<4><<<nblk, 256, 0, stream>>>(xin[l], WBF, bih, bhh,
                                                        Gb, M, N, K, N);

            lstm_recurrent_kernel<<<1, 1024, 0, stream>>>(Gb, Whh, xout[l],
                                                          /*reverse=*/d,
                                                          /*col_off=*/d * HID);
        }
    }

    // 3) emission head: EMI[T,48] = L2 @ W_out^T + b_out
    //    1 wave/block, NT=3: block covers 16 x 48; grid = 128
    {
        int M = T_LEN, N = KTAG, K = 2 * HID;
        int wn = N * K;
        cvt_bf16_kernel<<<(wn + 255) / 256, 256, 0, stream>>>(W_out, WBF, wn);
        int nblk = (M >> 4);
        gemm_wmma_bf16<3><<<nblk, 32, 0, stream>>>(L2, WBF, b_out, nullptr,
                                                   EMI, M, N, K, N);
    }

    // 4) CRF
    crf_kernel<<<1, 64, 0, stream>>>(EMI, trans, tags, (float*)d_out);
}
